// DeformableProjectionModule_28845000360118
// MI455X (gfx1250) — compile-verified, matrix operand order checked by reading, not run on hardware
//
#include <hip/hip_runtime.h>
#include <hip/hip_bf16.h>
#include <math.h>

// Problem constants (from reference)
#define BB    2
#define CCH   256
#define HH    64
#define WW    64
#define HWSZ  4096      // H*W
#define TT    29
#define TDIM  512
#define NHEAD 8
#define DHEAD 32
#define GG    4
#define CGRP  64
#define KPTS  9

typedef __attribute__((ext_vector_type(16))) _Float16 v16h;
typedef __attribute__((ext_vector_type(8)))  _Float16 v8h;
typedef __attribute__((ext_vector_type(8)))  float    v8f;

// ---------------------------------------------------------------------------
// fp32 -> f16 conversion (weights / activations)
// ---------------------------------------------------------------------------
__global__ __launch_bounds__(256) void f32_to_f16_kernel(const float* __restrict__ src,
                                                         _Float16* __restrict__ dst, int n) {
  int i = blockIdx.x * 256 + threadIdx.x;
  if (i < n) dst[i] = (_Float16)src[i];
}

// ---------------------------------------------------------------------------
// Text projection + K/V projection (tiny: B*T = 58 rows)
// block = 256 threads, one block per (b,t)
// ---------------------------------------------------------------------------
__global__ __launch_bounds__(256) void text_kv_kernel(
    const float* __restrict__ text_feat, const float* __restrict__ text_w,
    const float* __restrict__ text_b,
    const float* __restrict__ wk, const float* __restrict__ bk,
    const float* __restrict__ wv, const float* __restrict__ bv,
    float* __restrict__ kbuf, float* __restrict__ vbuf) {
  int bt = blockIdx.x;          // b*T + t
  int c  = threadIdx.x;
  const float* tf = text_feat + (size_t)bt * TDIM;
  const float* wr = text_w + (size_t)c * TDIM;
  float acc = text_b[c];
  for (int i = 0; i < TDIM; ++i) acc += tf[i] * wr[i];
  __shared__ float tl[CCH];
  tl[c] = acc;
  __syncthreads();
  const float* wkr = wk + (size_t)c * CCH;
  const float* wvr = wv + (size_t)c * CCH;
  float ak = bk[c], av = bv[c];
  for (int i = 0; i < CCH; ++i) { ak += tl[i] * wkr[i]; av += tl[i] * wvr[i]; }
  kbuf[(size_t)bt * CCH + c] = ak;
  vbuf[(size_t)bt * CCH + c] = av;
}

// ---------------------------------------------------------------------------
// LayerNorm over C of vseq (strided gather from NCHW visual feat), optional
// residual add, writes f16 row-major [row, C] for WMMA consumption.
// block = 256 threads = one row (b*HW + l)
// ---------------------------------------------------------------------------
__global__ __launch_bounds__(256) void ln_rows_kernel(
    const float* __restrict__ vis, const float* __restrict__ resid,
    const float* __restrict__ g, const float* __restrict__ beta,
    _Float16* __restrict__ outf16) {
  int row = blockIdx.x;                 // b*HW + l
  int b = row >> 12;                    // HW = 4096
  int l = row & (HWSZ - 1);
  int c = threadIdx.x;
  float x = vis[((size_t)b * CCH + c) * HWSZ + l];
  if (resid) x += resid[(size_t)row * CCH + c];
  __shared__ float red[CCH];
  red[c] = x;
  __syncthreads();
  for (int s = 128; s > 0; s >>= 1) { if (c < s) red[c] += red[c + s]; __syncthreads(); }
  float mean = red[0] * (1.0f / CCH);
  __syncthreads();
  float d = x - mean;
  red[c] = d * d;
  __syncthreads();
  for (int s = 128; s > 0; s >>= 1) { if (c < s) red[c] += red[c + s]; __syncthreads(); }
  float var = red[0] * (1.0f / CCH);
  float y = d * rsqrtf(var + 1e-5f) * g[c] + beta[c];
  outf16[(size_t)row * CCH + c] = (_Float16)y;
}

// ---------------------------------------------------------------------------
// WMMA GEMM: out[l,o] = sum_c A[l,c]*W[o,c] + bias[o]
// A: [M,256] f16 row-major, W: [256,256] f16 row-major, out: [M,256] f32
// Block: 256 threads = 8 waves. Block tile 128x64: each wave owns one 16-row
// M-subtile and sweeps ALL FOUR 16-col N-subtiles, so each A fragment is
// loaded from global once per K-step and reused by 4 WMMAs (4x less A
// traffic than a 2x4 wave map). K = 256 fully unrolled -> 32 v_wmma per wave.
// Weight tile (64 x 256 f16 = 32 KB) staged in LDS.
// ---------------------------------------------------------------------------
__global__ __launch_bounds__(256) void gemm_wmma_kernel(
    const _Float16* __restrict__ A, const _Float16* __restrict__ Wt,
    const float* __restrict__ bias, float* __restrict__ out) {
  constexpr int K = 256, N = 256, TILE_M = 128, TILE_N = 64;
  __shared__ _Float16 ldsB[TILE_N * K];           // 32 KB
  const int tid = threadIdx.x;
  const int bm = blockIdx.x;                      // M / 128
  const int bn = blockIdx.y;                      // N / 64

  // Stage 64 weight rows (bn*64 .. +64) x 256 f16 into LDS.
  {
    int row = tid >> 2, chunk = tid & 3;          // 64 rows x 4 chunks of 64 f16
    const v8h* s8 = (const v8h*)(Wt + (size_t)(bn * TILE_N + row) * K + chunk * 64);
    v8h* d8 = (v8h*)(ldsB + row * K + chunk * 64);
#pragma unroll
    for (int i = 0; i < 8; ++i) d8[i] = s8[i];
  }
  __syncthreads();

  const int wave = tid >> 5, lane = tid & 31;
  const int mrow = bm * TILE_M + wave * 16;       // wave -> M subtile
  const int lmod = lane & 15, lhalf = lane >> 4;

  const _Float16* arow = A + (size_t)(mrow + lmod) * K;
  __builtin_prefetch(arow + K, 0, 1);             // gfx1250 global_prefetch_b8

  v8f acc0 = {}, acc1 = {}, acc2 = {}, acc3 = {};
#pragma unroll
  for (int kb = 0; kb < K; kb += 32) {
    // 16-bit A/B fragment layout: lane-half h owns K subsets {h*8..h*8+7} and
    // {16+h*8..16+h*8+7} within the 32-wide K block (two 16B chunks).
    v8h a0 = *(const v8h*)(arow + kb + lhalf * 8);
    v8h a1 = *(const v8h*)(arow + kb + 16 + lhalf * 8);
    v16h af;
#pragma unroll
    for (int j = 0; j < 8; ++j) { af[j] = a0[j]; af[j + 8] = a1[j]; }
#pragma unroll
    for (int nt = 0; nt < 4; ++nt) {
      const _Float16* brow = ldsB + (size_t)(nt * 16 + lmod) * K;
      v8h b0 = *(const v8h*)(brow + kb + lhalf * 8);
      v8h b1 = *(const v8h*)(brow + kb + 16 + lhalf * 8);
      v16h bf;
#pragma unroll
      for (int j = 0; j < 8; ++j) { bf[j] = b0[j]; bf[j + 8] = b1[j]; }
      v8f c = (nt == 0) ? acc0 : (nt == 1) ? acc1 : (nt == 2) ? acc2 : acc3;
      c = __builtin_amdgcn_wmma_f32_16x16x32_f16(false, af, false, bf, (short)0, c, false, false);
      if (nt == 0) acc0 = c; else if (nt == 1) acc1 = c; else if (nt == 2) acc2 = c; else acc3 = c;
    }
  }

  // D layout: lane = N column (lane&15), VGPR r -> M = r + 8*(lane>>4)
#pragma unroll
  for (int nt = 0; nt < 4; ++nt) {
    const v8f acc = (nt == 0) ? acc0 : (nt == 1) ? acc1 : (nt == 2) ? acc2 : acc3;
    const int col = bn * TILE_N + nt * 16 + lmod;
    const float bv = bias ? bias[col] : 0.0f;
#pragma unroll
    for (int r = 0; r < 8; ++r) {
      int row = mrow + r + 8 * lhalf;
      out[(size_t)row * N + col] = acc[r] + bv;
    }
  }
}

// ---------------------------------------------------------------------------
// Cross-attention over T=29 text tokens; one thread per (b, head, l).
// K/V head slices staged in LDS. Writes attn output as f16 GEMM input.
// ---------------------------------------------------------------------------
__global__ __launch_bounds__(256) void attention_kernel(
    const float* __restrict__ qbuf, const float* __restrict__ kbuf,
    const float* __restrict__ vbuf, _Float16* __restrict__ aof16) {
  const int bh = blockIdx.y;            // b*NHEAD + head
  const int b = bh >> 3, head = bh & 7;
  const int l = blockIdx.x * 256 + threadIdx.x;
  __shared__ float ks[TT][DHEAD];
  __shared__ float vs[TT][DHEAD];
  for (int i = threadIdx.x; i < TT * DHEAD; i += 256) {
    int t = i >> 5, d = i & 31;
    ks[t][d] = kbuf[((size_t)b * TT + t) * CCH + head * DHEAD + d];
    vs[t][d] = vbuf[((size_t)b * TT + t) * CCH + head * DHEAD + d];
  }
  __syncthreads();
  const float* qp = qbuf + ((size_t)b * HWSZ + l) * CCH + head * DHEAD;
  float qr[DHEAD];
#pragma unroll
  for (int d = 0; d < DHEAD; ++d) qr[d] = qp[d];
  const float scale = 0.1767766952966369f;   // 1/sqrt(32)
  float logits[TT], mx = -1e30f;
  for (int t = 0; t < TT; ++t) {
    float s = 0.0f;
#pragma unroll
    for (int d = 0; d < DHEAD; ++d) s += qr[d] * ks[t][d];
    s *= scale;
    logits[t] = s;
    mx = fmaxf(mx, s);
  }
  float se = 0.0f;
  for (int t = 0; t < TT; ++t) { logits[t] = __expf(logits[t] - mx); se += logits[t]; }
  float inv = 1.0f / se;
  float o[DHEAD];
#pragma unroll
  for (int d = 0; d < DHEAD; ++d) o[d] = 0.0f;
  for (int t = 0; t < TT; ++t) {
    float w = logits[t] * inv;
#pragma unroll
    for (int d = 0; d < DHEAD; ++d) o[d] += w * vs[t][d];
  }
  _Float16* op = aof16 + ((size_t)b * HWSZ + l) * CCH + head * DHEAD;
#pragma unroll
  for (int d = 0; d < DHEAD; ++d) op[d] = (_Float16)o[d];
}

// ---------------------------------------------------------------------------
// Offset/mask projection: om[row, 0..107] = vseq2 . om_w[j] + om_b[j]
// ---------------------------------------------------------------------------
__global__ __launch_bounds__(128) void om_kernel(
    const _Float16* __restrict__ vseq2f16, const float* __restrict__ om_w,
    const float* __restrict__ om_b, float* __restrict__ om) {
  const int row = blockIdx.x;
  __shared__ float xs[CCH];
  for (int i = threadIdx.x; i < CCH; i += 128)
    xs[i] = (float)vseq2f16[(size_t)row * CCH + i];
  __syncthreads();
  const int j = threadIdx.x;
  if (j < GG * 3 * KPTS) {
    const float* wr = om_w + (size_t)j * CCH;
    float a = om_b[j];
    for (int i = 0; i < CCH; ++i) a += xs[i] * wr[i];
    om[(size_t)row * (GG * 3 * KPTS) + j] = a;
  }
}

// ---------------------------------------------------------------------------
// Deformable sampling: one block per (b,l), thread = (g, c) in 4x64.
// ---------------------------------------------------------------------------
__device__ __forceinline__ float dcn_tap(const float* __restrict__ val,
                                         int b, int y, int x, int gc) {
  if (y < 0 || y >= HH || x < 0 || x >= WW) return 0.0f;
  return val[((size_t)b * HWSZ + y * WW + x) * CCH + gc];
}

__global__ __launch_bounds__(256) void dcn_kernel(
    const float* __restrict__ val, const float* __restrict__ om,
    _Float16* __restrict__ dcnf16) {
  const int row = blockIdx.x;             // b*HW + l
  const int b = row >> 12;
  const int l = row & (HWSZ - 1);
  const int hh = l >> 6, wp = l & 63;
  __shared__ float oms[GG * 3 * KPTS];
  if (threadIdx.x < GG * 3 * KPTS)
    oms[threadIdx.x] = om[(size_t)row * (GG * 3 * KPTS) + threadIdx.x];
  __syncthreads();
  const int g = threadIdx.x >> 6, c = threadIdx.x & 63;
  const float* og = oms + g * (3 * KPTS);
  const int gc = g * CGRP + c;
  float acc = 0.0f;
#pragma unroll
  for (int k = 0; k < KPTS; ++k) {
    float px = (float)wp + (float)(k % 3 - 1) + og[2 * k];
    float py = (float)hh + (float)(k / 3 - 1) + og[2 * k + 1];
    float m = og[2 * KPTS + k];
    float x0f = floorf(px), y0f = floorf(py);
    float fx = px - x0f, fy = py - y0f;
    int x0 = (int)x0f, y0 = (int)y0f;
    float s = dcn_tap(val, b, y0,     x0,     gc) * (1.0f - fx) * (1.0f - fy)
            + dcn_tap(val, b, y0,     x0 + 1, gc) * fx * (1.0f - fy)
            + dcn_tap(val, b, y0 + 1, x0,     gc) * (1.0f - fx) * fy
            + dcn_tap(val, b, y0 + 1, x0 + 1, gc) * fx * fy;
    acc += s * m;
  }
  dcnf16[(size_t)row * CCH + gc] = (_Float16)acc;
}

// ---------------------------------------------------------------------------
// Exact GELU (erf form) with f16 output for the fuse GEMM
// ---------------------------------------------------------------------------
__global__ __launch_bounds__(256) void gelu_f16_kernel(
    const float* __restrict__ src, _Float16* __restrict__ dst, int n) {
  int i = blockIdx.x * 256 + threadIdx.x;
  if (i < n) {
    float x = src[i];
    dst[i] = (_Float16)(0.5f * x * (1.0f + erff(x * 0.7071067811865476f)));
  }
}

// ---------------------------------------------------------------------------
// Final: out[b,o,h,w] = visual[b,o,h,w] + fused[(b*HW+l), o]
// (fuse bias already applied inside the GEMM)
// ---------------------------------------------------------------------------
__global__ __launch_bounds__(256) void final_add_kernel(
    const float* __restrict__ vis, const float* __restrict__ fused,
    float* __restrict__ out) {
  int idx = blockIdx.x * 256 + threadIdx.x;   // over B*C*HW
  int l = idx & (HWSZ - 1);
  int rest = idx >> 12;
  int o = rest & (CCH - 1);
  int b = rest >> 8;
  out[idx] = vis[idx] + fused[((size_t)b * HWSZ + l) * CCH + o];
}

// ---------------------------------------------------------------------------
// Host launch
// ---------------------------------------------------------------------------
extern "C" void kernel_launch(void* const* d_in, const int* in_sizes, int n_in,
                              void* d_out, int out_size, void* d_ws, size_t ws_size,
                              hipStream_t stream) {
  const float* visual   = (const float*)d_in[0];
  const float* text_feat= (const float*)d_in[1];
  const float* text_w   = (const float*)d_in[2];
  const float* text_b   = (const float*)d_in[3];
  const float* wq       = (const float*)d_in[4];
  const float* bq       = (const float*)d_in[5];
  const float* wk       = (const float*)d_in[6];
  const float* bk       = (const float*)d_in[7];
  const float* wv       = (const float*)d_in[8];
  const float* bv       = (const float*)d_in[9];
  const float* attn_ow  = (const float*)d_in[10];
  const float* attn_ob  = (const float*)d_in[11];
  const float* ln1_g    = (const float*)d_in[12];
  const float* ln1_b    = (const float*)d_in[13];
  const float* ln2_g    = (const float*)d_in[14];
  const float* ln2_b    = (const float*)d_in[15];
  const float* val_w    = (const float*)d_in[16];
  const float* val_b    = (const float*)d_in[17];
  const float* om_w     = (const float*)d_in[18];
  const float* om_b     = (const float*)d_in[19];
  const float* dcn_ow   = (const float*)d_in[20];
  const float* dcn_ob   = (const float*)d_in[21];
  const float* fuse_w   = (const float*)d_in[22];
  const float* fuse_b   = (const float*)d_in[23];
  float* out = (float*)d_out;

  const int M = BB * HWSZ;                 // 8192 rows
  char* ws = (char*)d_ws;
  size_t off = 0;
  auto alloc = [&](size_t bytes) { size_t o = off; off = (off + bytes + 255) & ~(size_t)255; return o; };

  // f16 weight copies (5 x 256x256)
  const size_t wbytes = (size_t)CCH * CCH * sizeof(_Float16);
  _Float16* wq16   = (_Float16*)(ws + alloc(wbytes));
  _Float16* aow16  = (_Float16*)(ws + alloc(wbytes));
  _Float16* valw16 = (_Float16*)(ws + alloc(wbytes));
  _Float16* dcnw16 = (_Float16*)(ws + alloc(wbytes));
  _Float16* fusew16= (_Float16*)(ws + alloc(wbytes));
  float* kbuf = (float*)(ws + alloc((size_t)BB * TT * CCH * sizeof(float)));
  float* vbuf = (float*)(ws + alloc((size_t)BB * TT * CCH * sizeof(float)));
  // reused activation buffers
  _Float16* F = (_Float16*)(ws + alloc((size_t)M * CCH * sizeof(_Float16))); // 4 MB f16
  float*    X = (float*)   (ws + alloc((size_t)M * CCH * sizeof(float)));    // 8 MB f32
  float*    ombuf = (float*)(ws + alloc((size_t)M * GG * 3 * KPTS * sizeof(float)));
  (void)ws_size; (void)n_in; (void)in_sizes; (void)out_size;

  const int wn = CCH * CCH;
  dim3 cvt((wn + 255) / 256);
  f32_to_f16_kernel<<<cvt, 256, 0, stream>>>(wq,      wq16,    wn);
  f32_to_f16_kernel<<<cvt, 256, 0, stream>>>(attn_ow, aow16,   wn);
  f32_to_f16_kernel<<<cvt, 256, 0, stream>>>(val_w,   valw16,  wn);
  f32_to_f16_kernel<<<cvt, 256, 0, stream>>>(dcn_ow,  dcnw16,  wn);
  f32_to_f16_kernel<<<cvt, 256, 0, stream>>>(fuse_w,  fusew16, wn);

  // text -> k, v
  text_kv_kernel<<<BB * TT, 256, 0, stream>>>(text_feat, text_w, text_b,
                                              wk, bk, wv, bv, kbuf, vbuf);
  // LN1(vseq) -> F (f16)
  ln_rows_kernel<<<M, 256, 0, stream>>>(visual, nullptr, ln1_g, ln1_b, F);
  // q = F @ wq^T + bq -> X (f32)
  dim3 ggrid(M / 128, CCH / 64);
  gemm_wmma_kernel<<<ggrid, 256, 0, stream>>>(F, wq16, bq, X);
  // attention(q=X, k, v) -> F (f16)
  attention_kernel<<<dim3(HWSZ / 256, BB * NHEAD), 256, 0, stream>>>(X, kbuf, vbuf, F);
  // attn out proj: F @ attn_ow^T + attn_ob -> X
  gemm_wmma_kernel<<<ggrid, 256, 0, stream>>>(F, aow16, attn_ob, X);
  // LN2(vseq + X) -> F (f16)
  ln_rows_kernel<<<M, 256, 0, stream>>>(visual, X, ln2_g, ln2_b, F);
  // val = F @ val_w^T + val_b -> X
  gemm_wmma_kernel<<<ggrid, 256, 0, stream>>>(F, valw16, val_b, X);
  // om = F @ om_w^T + om_b -> ombuf   (must run before dcn overwrites F)
  om_kernel<<<M, 128, 0, stream>>>(F, om_w, om_b, ombuf);
  // deformable sampling: (X=val, ombuf) -> F (f16)
  dcn_kernel<<<M, 256, 0, stream>>>(X, ombuf, F);
  // dcn proj: F @ dcn_ow^T + dcn_ob -> X
  gemm_wmma_kernel<<<ggrid, 256, 0, stream>>>(F, dcnw16, dcn_ob, X);
  // gelu(X) -> F (f16)
  gelu_f16_kernel<<<(M * CCH) / 256, 256, 0, stream>>>(X, F, M * CCH);
  // fused = F @ fuse_w^T + fuse_b -> X
  gemm_wmma_kernel<<<ggrid, 256, 0, stream>>>(F, fusew16, fuse_b, X);
  // out = visual + fused (transposed back to NCHW)
  final_add_kernel<<<(BB * CCH * HWSZ) / 256, 256, 0, stream>>>(visual, X, out);
}
